// Detection_23785528885376
// MI455X (gfx1250) — compile-verified, matrix operand order checked by reading, not run on hardware
//
#include <hip/hip_runtime.h>
#include <stdint.h>

// ---------------------------------------------------------------------------
// Detection score for MI455X (gfx1250, wave32).
//
// Reference math collapses: nn_idx[0] == argmin_j dist(c_0, c_j) with
// first-occurrence tie-break (self-distance is exactly 0), so only ROW 0 of
// the NxN distance matrix matters. We compute that row exactly with
// v_wmma_f32_16x16x32_f16 (coords < 100 => f16 inputs exact, f32 accumulate
// exact), then a fused O(N*C) gamma pass (one wave32 per row, C==32 lanes),
// a deterministic per-batch sum-of-squares, and an in-place normalize.
// Total traffic ~2 MB reads / 64 KB writes -> launch-latency bound at
// 23.3 TB/s HBM.
// ---------------------------------------------------------------------------

typedef __attribute__((ext_vector_type(16))) _Float16 v16h;
typedef __attribute__((ext_vector_type(8)))  float    v8f;

#define N_PTS 8192
#define N_CH  32

// ws layout (16 bytes used): u32 pm[2] (packed argmin key per batch),
//                            f32 ss[2] (sum of gamma^2 per batch)

__global__ void init_ws_kernel(uint32_t* __restrict__ pm, int B) {
    int i = threadIdx.x;
    if (i < B) pm[i] = 0xFFFFFFFFu;
}

// --- Kernel A: row-0 nearest neighbor via WMMA ------------------------------
// One wave per 16-row tile. A = 16x32 f16 tile of coords (xyz in K=0..2,
// rest zero); B = 32x16 with every column = c0. D[m][n] = c_{row0+m} . c0.
// d2 = |c_m|^2 + |c0|^2 - 2*dot  (exact integers in f32).
// key = (d2 << 13) | idx : integer atomicMin gives min-d2 then min-idx,
// matching jnp.argmin first-occurrence tie-breaking. Order-independent.
__global__ __launch_bounds__(256) void nn0_kernel(const int* __restrict__ coords,
                                                  uint32_t* __restrict__ pm) {
    const int blocksPerBatch = (N_PTS / 16) / 8;   // 64
    int b    = blockIdx.x / blocksPerBatch;
    int wave = threadIdx.x >> 5;
    int lane = threadIdx.x & 31;
    int tile = (blockIdx.x % blocksPerBatch) * 8 + wave;
    int row0 = tile * 16;
    const int* cb = coords + (size_t)b * N_PTS * 3;

    // c0 (uniform -> scalar loads)
    float c0x = (float)cb[0], c0y = (float)cb[1], c0z = (float)cb[2];
    float sq0 = c0x * c0x + c0y * c0y + c0z * c0z;

    // Build A (rows = coord tile) and B (all 16 columns = c0).
    // 16-bit A layout: lanes 0-15 hold K=0..7 in elements 0..7 (elem e -> K=e),
    // lanes 16-31 hold K=8..15 -> keep those lanes zero so only K=0,1,2 live.
    v16h a = {};
    v16h bm = {};
    float sqr = 0.0f;
    if (lane < 16) {
        int r = row0 + lane;
        float x = (float)cb[r * 3 + 0];
        float y = (float)cb[r * 3 + 1];
        float z = (float)cb[r * 3 + 2];
        a[0] = (_Float16)x;  a[1] = (_Float16)y;  a[2] = (_Float16)z;
        bm[0] = (_Float16)c0x; bm[1] = (_Float16)c0y; bm[2] = (_Float16)c0z;
        sqr = x * x + y * y + z * z;
    }

    v8f acc = {};
    // (neg_a, A, neg_b, B, c_mod, C, reuse_a, reuse_b)
    acc = __builtin_amdgcn_wmma_f32_16x16x32_f16(
        false, a, false, bm, (short)0, acc, false, false);

    // C/D layout: lanes 0-15 hold rows m = vgpr (0..7); lanes 16-31 rows m+8.
    // sq of row (row0+m) lives in lane m (m=0..15): fetch via shuffles.
    int half = (lane >> 4) << 3;   // 0 for lanes 0-15, 8 for lanes 16-31
    float best_d2 = 1e30f;
    int   best_m  = 0;
#pragma unroll
    for (int m = 0; m < 8; ++m) {
        float sqm = __shfl(sqr, m + half, 32);
        float d2  = fmaxf(sqm + sq0 - 2.0f * acc[m], 0.0f);
        if (d2 < best_d2) { best_d2 = d2; best_m = m; }  // strict '<' keeps first idx
    }
    int gidx = row0 + best_m + half;                  // 0..8191, fits 13 bits
    uint32_t key = (((uint32_t)(best_d2 + 0.5f)) << 13) | (uint32_t)gidx;

    // wave-min of packed keys, then one atomicMin per wave
#pragma unroll
    for (int s = 16; s >= 1; s >>= 1) {
        uint32_t o = (uint32_t)__shfl_xor((int)key, s, 32);
        key = (o < key) ? o : key;
    }
    if (lane == 0) atomicMin(&pm[b], key);
}

// --- Kernel B: fused relu / rowmax / alpha*beta / gamma ---------------------
// One wave32 per row; lane c handles channel c (C == 32 == wave width).
__global__ __launch_bounds__(256) void gamma_kernel(const float* __restrict__ feat,
                                                    const uint32_t* __restrict__ pm,
                                                    float* __restrict__ gamma_out) {
    const int blocksPerBatch = N_PTS / 8;            // 1024
    int b    = blockIdx.x / blocksPerBatch;
    int wave = threadIdx.x >> 5;
    int lane = threadIdx.x & 31;
    int row  = (blockIdx.x % blocksPerBatch) * 8 + wave;

    int nn0 = (int)(pm[b] & 0x1FFFu);
    const float* fb = feat + (size_t)b * N_PTS * N_CH;

    float f0 = fmaxf(fb[(size_t)nn0 * N_CH + lane], 0.0f);   // relu(f[nn0])
    float fc = fmaxf(fb[(size_t)row * N_CH + lane], 0.0f);   // relu(f[row])

    // row max across the 32 channels
    float rmax = fc;
#pragma unroll
    for (int s = 16; s >= 1; s >>= 1) rmax = fmaxf(rmax, __shfl_xor(rmax, s, 32));

    // alpha * beta = exp(fc - f0) * fc / rmax ; gamma = max over channels
    float val = __expf(fc - f0) * fc / rmax;
#pragma unroll
    for (int s = 16; s >= 1; s >>= 1) val = fmaxf(val, __shfl_xor(val, s, 32));

    if (lane == 0) gamma_out[(size_t)b * N_PTS + row] = val;
}

// --- Kernel C: deterministic per-batch sum(gamma^2) (fixed-order LDS tree) --
__global__ __launch_bounds__(256) void sumsq_kernel(const float* __restrict__ gamma,
                                                    float* __restrict__ ss) {
    __shared__ float red[256];
    int b = blockIdx.x;
    const float* g = gamma + (size_t)b * N_PTS;
    float s = 0.0f;
    for (int i = threadIdx.x; i < N_PTS; i += 256) { float v = g[i]; s += v * v; }
    red[threadIdx.x] = s;
    __syncthreads();
#pragma unroll
    for (int w = 128; w >= 1; w >>= 1) {
        if (threadIdx.x < w) red[threadIdx.x] += red[threadIdx.x + w];
        __syncthreads();
    }
    if (threadIdx.x == 0) ss[b] = red[0];
}

// --- Kernel D: normalize in place -------------------------------------------
__global__ __launch_bounds__(256) void normalize_kernel(float* __restrict__ out,
                                                        const float* __restrict__ ss) {
    int i = blockIdx.x * blockDim.x + threadIdx.x;
    int b = i / N_PTS;
    out[i] = out[i] * rsqrtf(ss[b]);
}

extern "C" void kernel_launch(void* const* d_in, const int* in_sizes, int n_in,
                              void* d_out, int out_size, void* d_ws, size_t ws_size,
                              hipStream_t stream) {
    const int*   coords = (const int*)d_in[0];
    const float* feat   = (const float*)d_in[1];
    int B = in_sizes[1] / (N_PTS * N_CH);            // 2 for the reference shapes

    float*    out = (float*)d_out;                   // gamma staged here, normalized in place
    uint32_t* pm  = (uint32_t*)d_ws;                 // packed argmin keys
    float*    ss  = (float*)d_ws + 2;                // sum of squares

    init_ws_kernel<<<1, 32, 0, stream>>>(pm, B);
    nn0_kernel<<<B * 64, 256, 0, stream>>>(coords, pm);
    gamma_kernel<<<B * 1024, 256, 0, stream>>>(feat, pm, out);
    sumsq_kernel<<<B, 256, 0, stream>>>(out, ss);
    normalize_kernel<<<(B * N_PTS + 255) / 256, 256, 0, stream>>>(out, ss);
}
